// GraphNetwork_39161511805167
// MI455X (gfx1250) — matrix operand order repaired and checked by reading, not compiled, not run on hardware
//
#include <hip/hip_runtime.h>

// ---------------------------------------------------------------------------
// GraphNetwork on MI455X (gfx1250, wave32, WMMA).
//
// conv(feat, W) recast as GEMM: t[e, b*Cin+c] = Bp[e,b]*feat[j(e),c] built in
// LDS, msg = t @ W_flat. Dominant layers: [800000 x 1024] x [1024 x 64] f16
// GEMMs -> v_wmma_f32_16x16x32_f16, f32 accum. Weights pre-transposed to
// [Ncol][K] and preloaded into VGPRs (tile-invariant), so the steady-state
// inner loop is 2x ds_load_b128 + v_wmma. Feature gather uses
// GLOBAL_LOAD_ASYNC_TO_LDS_B64 (ASYNCcnt) when the builtin is available.
// Scatter-add via global_atomic_add_f32 (nodes live in the 192MB L2).
// ---------------------------------------------------------------------------

typedef __attribute__((ext_vector_type(16))) _Float16 v16h;
typedef __attribute__((ext_vector_type(8)))  _Float16 v8h;
typedef __attribute__((ext_vector_type(4)))  _Float16 v4h;
typedef __attribute__((ext_vector_type(8)))  float    v8f;
typedef __attribute__((ext_vector_type(2)))  int      v2i;

#if defined(__has_builtin)
#  if __has_builtin(__builtin_amdgcn_global_load_async_to_lds_b64)
#    define ASYNC_GATHER 1
#  endif
#endif
#ifndef ASYNC_GATHER
#  define ASYNC_GATHER 0
#endif

__device__ __forceinline__ void copy_b64_to_lds(const void* g, void* l) {
#if ASYNC_GATHER
  // Builtin prototype (probe-confirmed this round): (v2i AS1*, v2i AS3*, Ii, Ii).
  // AS1 = 64-bit global address; AS3 = low 32 bits of the generic LDS address.
  __builtin_amdgcn_global_load_async_to_lds_b64(
      (__attribute__((address_space(1))) v2i*)(uintptr_t)g,
      (__attribute__((address_space(3))) v2i*)(unsigned)(uintptr_t)l, 0, 0);
#else
  *(unsigned long long*)l = *(const unsigned long long*)g;
#endif
}

__device__ __forceinline__ void gather_wait() {
#if ASYNC_GATHER
#  if __has_builtin(__builtin_amdgcn_s_wait_asynccnt)
  __builtin_amdgcn_s_wait_asynccnt(0);
#  else
  asm volatile("s_wait_asynccnt 0x0" ::: "memory");
#  endif
#endif
}

__device__ __forceinline__ v16h cat16(v8h lo, v8h hi) {
  v16h r;
#pragma unroll
  for (int i = 0; i < 8; ++i) { r[i] = lo[i]; r[8 + i] = hi[i]; }
  return r;
}

// ---- hat-basis outer product per edge: Bp[e, bx*4+by] ---------------------
__global__ void basis_kernel(const float* __restrict__ edge_attr,
                             _Float16* __restrict__ bp, int n_edges)
{
  int e = blockIdx.x * blockDim.x + threadIdx.x;
  if (e >= n_edges) return;
  float px = edge_attr[2 * e], py = edge_attr[2 * e + 1];
  const float s = 1.5f;  // (N_BASIS-1)/2
  float bx[4], by[4];
#pragma unroll
  for (int i = 0; i < 4; ++i) {
    float c = -1.0f + (2.0f / 3.0f) * (float)i;
    bx[i] = fmaxf(0.0f, 1.0f - fabsf(px - c) * s);
    by[i] = fmaxf(0.0f, 1.0f - fabsf(py - c) * s);
  }
  _Float16* o = bp + (size_t)e * 16;
#pragma unroll
  for (int i = 0; i < 4; ++i)
#pragma unroll
    for (int j = 0; j < 4; ++j)
      o[i * 4 + j] = (_Float16)(bx[i] * by[j]);
}

// ---- W [K=16*Cin, Cout] f32 -> wt [Npad][K] f16 (transposed, zero-padded) -
__global__ void wconvert_kernel(const float* __restrict__ W, _Float16* __restrict__ wt,
                                int K, int cout, int npad)
{
  int i = blockIdx.x * blockDim.x + threadIdx.x;
  if (i >= npad * K) return;
  int n = i / K, k = i % K;
  wt[i] = (n < cout) ? (_Float16)W[(size_t)k * cout + n] : (_Float16)0.0f;
}

// ---- f32 -> f16 (optionally ReLU'd) feature buffer ------------------------
__global__ void f32_to_f16_kernel(const float* __restrict__ in, _Float16* __restrict__ out,
                                  long n, int do_relu)
{
  long i = (long)blockIdx.x * blockDim.x + threadIdx.x;
  if (i >= n) return;
  float v = in[i];
  if (do_relu) v = fmaxf(v, 0.0f);
  out[i] = (_Float16)v;
}

// ---- WMMA edge-conv: 16 edges/tile, NT col tiles x KSPLIT k-partitions ----
template <int CIN, int NT, int KSPLIT>
__global__ __launch_bounds__(NT* KSPLIT * 32)
void conv_wmma_kernel(const _Float16* __restrict__ feat,   // [N, CIN] f16
                      const _Float16* __restrict__ bp,     // [E, 16] f16
                      const _Float16* __restrict__ wt,     // [NT*16][K] transposed
                      const int* __restrict__ edge_i,
                      const int* __restrict__ edge_j,
                      float* __restrict__ acc,             // [N, cout] f32
                      int n_edges, int cout, int tiles_per_block)
{
  constexpr int K      = 16 * CIN;
  constexpr int NTH    = NT * KSPLIT * 32;
  constexpr int KC     = K / KSPLIT;      // K range per wave (>= 32)
  constexpr int KSTEPS = KC / 32;

  __shared__ alignas(16) _Float16 t_lds[16 * K];    // basis-weighted features
  __shared__ alignas(16) _Float16 f_lds[16 * CIN];  // gathered features
  __shared__ alignas(16) _Float16 b_lds[16 * 16];   // Bp tile
  __shared__ alignas(16) float    stage[(KSPLIT > 1) ? NT * (KSPLIT - 1) * 256 : 1];
  __shared__ int ei_lds[16];

  const int tid   = threadIdx.x;
  const int wave  = tid >> 5;
  const int lane  = tid & 31;
  const int ntile = wave % NT;
  const int kp    = wave / NT;
  const int ncol0 = ntile * 16;
  const int kbase = kp * KC;
  const int hi    = lane >> 4;
  const int row   = lane & 15;

  // ---- preload this wave's B fragments into VGPRs (tile-invariant) --------
  const _Float16* wrow = wt + (size_t)(ncol0 + row) * K + (hi ? 8 : 0);
  v16h breg[KSTEPS];
#pragma unroll
  for (int ks = 0; ks < KSTEPS; ++ks) {
    const int k0 = kbase + ks * 32;
    breg[ks] = cat16(*(const v8h*)(wrow + k0), *(const v8h*)(wrow + k0 + 16));
  }

  const int ntiles = (n_edges + 15) >> 4;
  for (int t = 0; t < tiles_per_block; ++t) {
    const int tile = blockIdx.x * tiles_per_block + t;   // uniform across block
    if (tile >= ntiles) break;
    const int e0 = tile << 4;

    __syncthreads();  // previous tile's LDS fully consumed

    if (tid < 16) {
      int e = e0 + tid;
      ei_lds[tid] = (e < n_edges) ? edge_i[e] : 0;
    }
    // gather features: one 8B async copy per chunk (per-lane addr -> LDS)
    for (int i = tid; i < 16 * (CIN / 4); i += NTH) {
      int e = i / (CIN / 4), c4 = i % (CIN / 4);
      int eg = e0 + e;
      if (eg < n_edges) {
        int j = edge_j[eg];
        copy_b64_to_lds(feat + (size_t)j * CIN + c4 * 4, &f_lds[i * 4]);
      }
    }
    // Bp tile (16x16 f16 = 64 x 8B chunks, contiguous)
    for (int i = tid; i < 64; i += NTH) {
      if (e0 + (i >> 2) < n_edges)
        copy_b64_to_lds(bp + (size_t)e0 * 16 + i * 4, &b_lds[i * 4]);
    }
    gather_wait();
    __syncthreads();

    // t[e][k] = Bp[e][k/CIN] * feat[e][k%CIN]; 4-f16 chunks never straddle a
    // basis boundary (CIN % 4 == 0) -> packed f16 multiplies.
    for (int i = tid; i < 16 * (K / 4); i += NTH) {
      int e  = i / (K / 4), kc = i % (K / 4);
      int b  = (kc * 4) / CIN;
      int c4 = kc % (CIN / 4);
      _Float16 s = b_lds[(e << 4) + b];
      v4h fv = ((const v4h*)f_lds)[e * (CIN / 4) + c4];
      v4h tv;
#pragma unroll
      for (int q = 0; q < 4; ++q) tv[q] = (_Float16)(s * fv[q]);
      ((v4h*)t_lds)[i] = tv;
    }
    __syncthreads();

    // ---- GEMM: C[16x16] += A(t)[16x32] x B(w)[32x16] over this wave's K ---
    v8f c = {};
#pragma unroll
    for (int ks = 0; ks < KSTEPS; ++ks) {
      const int k0 = kbase + ks * 32;
      const _Float16* ap = &t_lds[row * K + k0 + (hi ? 8 : 0)];
      v16h a = cat16(*(const v8h*)ap, *(const v8h*)(ap + 16));
      c = __builtin_amdgcn_wmma_f32_16x16x32_f16(false, a, false, breg[ks],
                                                 (short)0, c, false, false);
    }

    const int rbase = hi * 8;
    if constexpr (KSPLIT > 1) {
      // reduce K partials through LDS; only kp==0 waves touch global atomics
      if (kp > 0) {
        float* sp = &stage[(ntile * (KSPLIT - 1) + (kp - 1)) * 256];
#pragma unroll
        for (int r = 0; r < 8; ++r) sp[(rbase + r) * 16 + row] = c[r];
      }
      __syncthreads();
      if (kp == 0) {
#pragma unroll
        for (int q = 1; q < KSPLIT; ++q) {
          const float* sp = &stage[(ntile * (KSPLIT - 1) + (q - 1)) * 256];
#pragma unroll
          for (int r = 0; r < 8; ++r) c[r] += sp[(rbase + r) * 16 + row];
        }
        const int n = ncol0 + row;
        if (n < cout) {
#pragma unroll
          for (int r = 0; r < 8; ++r) {
            int m = rbase + r;  // C layout: VGPR r -> M = r (+8 for hi lanes)
            if (e0 + m < n_edges)
              atomicAdd(&acc[(size_t)ei_lds[m] * cout + n], c[r]);
          }
        }
      }
    } else {
      const int n = ncol0 + row;
      if (n < cout) {
#pragma unroll
        for (int r = 0; r < 8; ++r) {
          int m = rbase + r;
          if (e0 + m < n_edges)
            atomicAdd(&acc[(size_t)ei_lds[m] * cout + n], c[r]);
        }
      }
    }
  }
}

// ---- layer 0 combine: ans[:, :32] = x@fw0+fb0 ; ans[:, 32:] = conv0 -------
__global__ void layer0_kernel(const float* __restrict__ x, const float* __restrict__ cacc,
                              const float* __restrict__ fw, const float* __restrict__ fb,
                              float* __restrict__ ans, int n_nodes)
{
  int i = blockIdx.x * blockDim.x + threadIdx.x;
  if (i >= n_nodes * 64) return;
  int n = i >> 6, d = i & 63;
  float v;
  if (d < 32) {
    v = fb[d];
#pragma unroll
    for (int c = 0; c < 4; ++c) v = fmaf(x[n * 4 + c], fw[c * 32 + d], v);
  } else {
    v = cacc[n * 32 + (d - 32)];
  }
  ans[i] = v;
}

// ---- post: out = conv_acc + relu(ans_prev)@fw + fb (+ ans_prev) , *scale --
template <int CIN, int COUT, bool RESID>
__global__ void post_kernel(const float* __restrict__ ans_prev,
                            const float* __restrict__ cacc,
                            const float* __restrict__ fw, const float* __restrict__ fb,
                            float* __restrict__ ans_out, float scale, int n_nodes)
{
  int i = blockIdx.x * blockDim.x + threadIdx.x;
  if (i >= n_nodes * COUT) return;
  int n = i / COUT, d = i % COUT;
  float v = fb[d] + cacc[i];
  const float* ap = ans_prev + (size_t)n * CIN;
  for (int c = 0; c < CIN; ++c) v = fmaf(fmaxf(ap[c], 0.0f), fw[c * COUT + d], v);
  if (RESID) v += ans_prev[(size_t)n * COUT + d];  // CIN==COUT on residual layer
  ans_out[i] = v * scale;
}

extern "C" void kernel_launch(void* const* d_in, const int* in_sizes, int n_in,
                              void* d_out, int out_size, void* d_ws, size_t ws_size,
                              hipStream_t stream)
{
  const float* x         = (const float*)d_in[0];
  const int*   edge_i    = (const int*)d_in[1];
  const int*   edge_j    = (const int*)d_in[2];
  const float* edge_attr = (const float*)d_in[3];
  const float* W[4]; const float* fw[4]; const float* fb[4];
  for (int l = 0; l < 4; ++l) {
    W[l]  = (const float*)d_in[4 + 3 * l];
    fw[l] = (const float*)d_in[5 + 3 * l];
    fb[l] = (const float*)d_in[6 + 3 * l];
  }
  float* out = (float*)d_out;

  const int n_nodes = in_sizes[0] / 4;
  const int n_edges = in_sizes[1];

  // workspace carve-out (256B aligned)
  char* p = (char*)d_ws;
  auto take = [&](size_t bytes) { char* q = p; p += (bytes + 255) & ~(size_t)255; return q; };
  _Float16* bp     = (_Float16*)take((size_t)n_edges * 16 * 2);
  _Float16* feat16 = (_Float16*)take((size_t)n_nodes * 64 * 2);
  float*    ansA   = (float*)take((size_t)n_nodes * 64 * 4);
  float*    ansB   = (float*)take((size_t)n_nodes * 64 * 4);
  float*    cacc   = (float*)take((size_t)n_nodes * 64 * 4);
  _Float16* wt0    = (_Float16*)take((size_t)32 * 64 * 2);
  _Float16* wt1    = (_Float16*)take((size_t)64 * 1024 * 2);
  _Float16* wt2    = (_Float16*)take((size_t)64 * 1024 * 2);
  _Float16* wt3    = (_Float16*)take((size_t)16 * 1024 * 2);
  (void)ws_size; (void)n_in; (void)out_size;

  const int ntiles      = (n_edges + 15) / 16;
  const int TPB         = 8;  // edge tiles per workgroup
  const int conv_blocks = (ntiles + TPB - 1) / TPB;

  basis_kernel<<<(n_edges + 255) / 256, 256, 0, stream>>>(edge_attr, bp, n_edges);
  wconvert_kernel<<<(32 * 64 + 255) / 256, 256, 0, stream>>>(W[0], wt0, 64, 32, 32);
  wconvert_kernel<<<(64 * 1024 + 255) / 256, 256, 0, stream>>>(W[1], wt1, 1024, 64, 64);
  wconvert_kernel<<<(64 * 1024 + 255) / 256, 256, 0, stream>>>(W[2], wt2, 1024, 64, 64);
  wconvert_kernel<<<(16 * 1024 + 255) / 256, 256, 0, stream>>>(W[3], wt3, 1024, 2, 16);

  // ---- layer 0: concat(fc(x), conv(x)) ------------------------------------
  f32_to_f16_kernel<<<(int)(((long)n_nodes * 4 + 255) / 256), 256, 0, stream>>>(
      x, feat16, (long)n_nodes * 4, 0);
  (void)hipMemsetAsync(cacc, 0, (size_t)n_nodes * 32 * 4, stream);
  conv_wmma_kernel<4, 2, 2><<<conv_blocks, 128, 0, stream>>>(
      feat16, bp, wt0, edge_i, edge_j, cacc, n_edges, 32, TPB);
  layer0_kernel<<<(int)(((long)n_nodes * 64 + 255) / 256), 256, 0, stream>>>(
      x, cacc, fw[0], fb[0], ansA, n_nodes);

  // ---- layer 1 ------------------------------------------------------------
  f32_to_f16_kernel<<<(int)(((long)n_nodes * 64 + 255) / 256), 256, 0, stream>>>(
      ansA, feat16, (long)n_nodes * 64, 1);
  (void)hipMemsetAsync(cacc, 0, (size_t)n_nodes * 64 * 4, stream);
  conv_wmma_kernel<64, 4, 2><<<conv_blocks, 256, 0, stream>>>(
      feat16, bp, wt1, edge_i, edge_j, cacc, n_edges, 64, TPB);
  post_kernel<64, 64, false><<<(int)(((long)n_nodes * 64 + 255) / 256), 256, 0, stream>>>(
      ansA, cacc, fw[1], fb[1], ansB, 1.0f, n_nodes);

  // ---- layer 2 (residual) -------------------------------------------------
  f32_to_f16_kernel<<<(int)(((long)n_nodes * 64 + 255) / 256), 256, 0, stream>>>(
      ansB, feat16, (long)n_nodes * 64, 1);
  (void)hipMemsetAsync(cacc, 0, (size_t)n_nodes * 64 * 4, stream);
  conv_wmma_kernel<64, 4, 2><<<conv_blocks, 256, 0, stream>>>(
      feat16, bp, wt2, edge_i, edge_j, cacc, n_edges, 64, TPB);
  post_kernel<64, 64, true><<<(int)(((long)n_nodes * 64 + 255) / 256), 256, 0, stream>>>(
      ansB, cacc, fw[2], fb[2], ansA, 1.0f, n_nodes);

  // ---- layer 3 (output, cout=2, K split across 8 waves) -------------------
  f32_to_f16_kernel<<<(int)(((long)n_nodes * 64 + 255) / 256), 256, 0, stream>>>(
      ansA, feat16, (long)n_nodes * 64, 1);
  (void)hipMemsetAsync(cacc, 0, (size_t)n_nodes * 2 * 4, stream);
  conv_wmma_kernel<64, 1, 8><<<conv_blocks, 256, 0, stream>>>(
      feat16, bp, wt3, edge_i, edge_j, cacc, n_edges, 2, TPB);
  post_kernel<64, 2, false><<<(int)(((long)n_nodes * 2 + 255) / 256), 256, 0, stream>>>(
      ansA, cacc, fw[3], fb[3], out, 1.0f / 128.0f, n_nodes);
}
